// RoBERTaBlock_81544249081992
// MI455X (gfx1250) — matrix operand order repaired and checked by reading, not compiled, not run on hardware
//
#include <hip/hip_runtime.h>
#include <hip/hip_bf16.h>
#include <math.h>

// ---------------------------------------------------------------------------
// Types for CDNA5 WMMA (wave32): bf16 operands, f32 accumulators.
// ---------------------------------------------------------------------------
typedef __bf16 bf16;
typedef __bf16 bf16x4  __attribute__((ext_vector_type(4)));
typedef __bf16 bf16x8  __attribute__((ext_vector_type(8)));
typedef __bf16 bf16x16 __attribute__((ext_vector_type(16)));
typedef float  floatx8 __attribute__((ext_vector_type(8)));

#define WMMA_BF16(a, b, c)                                                    \
  __builtin_amdgcn_wmma_f32_16x16x32_bf16(false, (a), false, (b), (short)0,   \
                                          (c), false, false)

// Async global -> LDS copy, 16 bytes per lane (CDNA5, tracked by ASYNCcnt).
// GV addressing: LDS byte offset comes from the vdst VGPR; global address from a 64-bit VGPR.
static __device__ __forceinline__ void async_copy_b128(unsigned lds_off,
                                                       const void* gaddr) {
  asm volatile("global_load_async_to_lds_b128 %0, %1, off"
               :
               : "v"(lds_off), "v"(gaddr)
               : "memory");
}
static __device__ __forceinline__ void wait_asynccnt0() {
  asm volatile("s_wait_asynccnt 0x0" ::: "memory");
}
// Generic->LDS-offset: low 32 bits of an AMDGPU generic LDS address are the
// byte offset from the workgroup LDS base (aperture in the high bits).
static __device__ __forceinline__ unsigned lds_offset_of(const void* p) {
  return (unsigned)(unsigned long long)p;
}

// Build a 16-element bf16 fragment from two contiguous 8-element (16-byte)
// LDS reads -> lowers to ds_load_b128 pairs.
static __device__ __forceinline__ bf16x16 frag_from(const bf16* p0,
                                                    const bf16* p1) {
  bf16x8 lo = *reinterpret_cast<const bf16x8*>(p0);
  bf16x8 hi = *reinterpret_cast<const bf16x8*>(p1);
  bf16x16 r;
#pragma unroll
  for (int i = 0; i < 8; ++i) { r[i] = lo[i]; r[i + 8] = hi[i]; }
  return r;
}

static __device__ __forceinline__ float gelu_tanh(float x) {
  const float c = 0.7978845608028654f;  // sqrt(2/pi)
  float x3 = x * x * x;
  return 0.5f * x * (1.f + tanhf(c * (x + 0.044715f * x3)));
}

// ---------------------------------------------------------------------------
// One-time weight prep: fp32 [K][N] -> bf16 transposed [N][K] so GEMM
// B-fragments are contiguous K runs per column (and async-copyable).
// ---------------------------------------------------------------------------
__global__ void castT_f32_bf16(const float* __restrict__ W,
                               bf16* __restrict__ Wt, int K, int N) {
  size_t i = (size_t)blockIdx.x * blockDim.x + threadIdx.x;
  size_t total = (size_t)K * N;
  if (i >= total) return;
  int kk = (int)(i / N);
  int nn = (int)(i % N);
  Wt[(size_t)nn * K + kk] = (bf16)W[i];
}

// ---------------------------------------------------------------------------
// Tiled bf16 WMMA GEMM:  C[M,N] = epilogue(A[M,K] @ B[K,N] + bias [+ resid])
// A: fp32 row-major (cast to bf16 through VGPRs). Bt: bf16 [N][K], moved
// global->LDS with async-to-LDS copies. Double-buffered LDS, L2 prefetch of
// A two K-steps ahead. Block tile 128x128, BK=32; 8 waves x (32x64) each.
// ---------------------------------------------------------------------------
template <bool GELU_ACT, bool RESID>
__global__ __launch_bounds__(256, 2) void gemm_bf16_wmma(
    const float* __restrict__ A, const bf16* __restrict__ Bt,
    const float* __restrict__ bias, const float* __restrict__ resid,
    float* __restrict__ C, int M, int N, int K) {
  constexpr int BM = 128, BN = 128, BK = 32, LDP = BK + 8;  // 80B rows (16B-aligned)
  __shared__ bf16 sA[2][BM][LDP];
  __shared__ bf16 sB[2][BN][LDP];
  constexpr unsigned bufAsz = BM * LDP * 2;
  constexpr unsigned bufBsz = BN * LDP * 2;

  const int m0 = blockIdx.y * BM;
  const int n0 = blockIdx.x * BN;
  const int t = threadIdx.x;
  const int wave = t >> 5, lane = t & 31;
  const int half = lane >> 4, r = lane & 15;
  const int wm0 = (wave & 3) * 32;   // 4 waves along M
  const int wn0 = (wave >> 2) * 64;  // 2 waves along N
  const int kOffA = half * 8;        // A frag K-offset per ISA layout
  const int kOffB = half * 8;        // B frag K-offset per ISA layout

  const int ar = t >> 3;          // A stage: 32 rows/pass x 4
  const int ac4 = (t & 7) << 2;   // float4 column
  const int br = t >> 2;          // B stage: 64 rows/pass x 2
  const int bc8 = (t & 3) << 3;   // 8 bf16 (16B) column

  const unsigned sA0 = lds_offset_of(&sA[0][0][0]);
  const unsigned sB0 = lds_offset_of(&sB[0][0][0]);
  (void)sA0;

  floatx8 acc[2][4];
#pragma unroll
  for (int mi = 0; mi < 2; ++mi)
#pragma unroll
    for (int ni = 0; ni < 4; ++ni)
#pragma unroll
      for (int e = 0; e < 8; ++e) acc[mi][ni][e] = 0.f;

  // Stage helpers -----------------------------------------------------------
  auto issueB = [&](int k0, int buf) {  // async global->LDS, no VGPR data
#pragma unroll
    for (int p = 0; p < 2; ++p) {
      int n = br + p * 64;
      async_copy_b128(sB0 + (unsigned)buf * bufBsz + (unsigned)(n * LDP + bc8) * 2,
                      Bt + (size_t)(n0 + n) * K + (k0 + bc8));
    }
  };
  float4 regA[4];
  auto loadA = [&](int k0) {
#pragma unroll
    for (int p = 0; p < 4; ++p) {
      int row = ar + p * 32;
      regA[p] = *reinterpret_cast<const float4*>(A + (size_t)(m0 + row) * K +
                                                 (k0 + ac4));
    }
  };
  auto storeA = [&](int buf) {  // fp32 -> bf16, VGPR -> LDS
#pragma unroll
    for (int p = 0; p < 4; ++p) {
      int row = ar + p * 32;
      bf16x4 h = {(bf16)regA[p].x, (bf16)regA[p].y, (bf16)regA[p].z,
                  (bf16)regA[p].w};
      *reinterpret_cast<bf16x4*>(&sA[buf][row][ac4]) = h;
    }
  };

  const int nkt = K / BK;
  issueB(0, 0);
  loadA(0);

  for (int kt = 0; kt < nkt; ++kt) {
    const int buf = kt & 1;
    storeA(buf);        // commit this tile's A (s_wait_dscnt before barrier)
    wait_asynccnt0();   // this tile's async B copies done
    __syncthreads();    // tile `buf` visible to all waves

    if (kt + 1 < nkt) {               // overlap next tile with compute
      issueB((kt + 1) * BK, buf ^ 1); // async B copies
      loadA((kt + 1) * BK);           // A f32 into regs
      if (kt + 2 < nkt)               // pull A two steps ahead into L2
        __builtin_prefetch(A + (size_t)(m0 + ar) * K + (kt + 2) * BK + ac4, 0, 1);
    }

    bf16x16 aF[2], bF[4];
#pragma unroll
    for (int mi = 0; mi < 2; ++mi) {
      const bf16* rp = &sA[buf][wm0 + mi * 16 + r][0];
      aF[mi] = frag_from(rp + kOffA, rp + 16 + kOffA);
    }
#pragma unroll
    for (int ni = 0; ni < 4; ++ni) {
      const bf16* rp = &sB[buf][wn0 + ni * 16 + r][0];
      bF[ni] = frag_from(rp + kOffB, rp + 16 + kOffB);
    }
#pragma unroll
    for (int mi = 0; mi < 2; ++mi)
#pragma unroll
      for (int ni = 0; ni < 4; ++ni)
        acc[mi][ni] = WMMA_BF16(aF[mi], bF[ni], acc[mi][ni]);
    // No trailing barrier: the next iteration only writes the other buffer,
    // and its top barrier orders all cross-wave reuse.
  }

  // Epilogue: bias (+ residual) (+ GELU). C layout: VGPR e -> row e+half*8,
  // lane%16 -> column.
#pragma unroll
  for (int mi = 0; mi < 2; ++mi)
#pragma unroll
    for (int ni = 0; ni < 4; ++ni) {
      int col = n0 + wn0 + ni * 16 + r;
      float bv = bias[col];
#pragma unroll
      for (int e = 0; e < 8; ++e) {
        int row = m0 + wm0 + mi * 16 + half * 8 + e;
        float v = acc[mi][ni][e] + bv;
        if (RESID) v += resid[(size_t)row * N + col];
        if (GELU_ACT) v = gelu_tanh(v);
        C[(size_t)row * N + col] = v;
      }
    }
}

// ---------------------------------------------------------------------------
// Flash attention, one (batch, head) per blockIdx.y, 128 query rows per block
// (8 waves x 16 rows). K/V staged in LDS in 64-key tiles (f32->bf16 + V
// transpose, so the VGPR staging path is required); online softmax with
// shfl_xor row reductions; P re-laid-out through per-wave LDS for the P@V
// WMMA A-fragments. d_k = 64.
// ---------------------------------------------------------------------------
__global__ __launch_bounds__(256, 2) void flash_attn(
    const float* __restrict__ Q, const float* __restrict__ Kv,
    const float* __restrict__ V, float* __restrict__ O, int Bn, int S, int H,
    int heads, int dk) {
  __shared__ bf16 sK[64][72];       // [key][d]     rows 144B (16B-aligned)
  __shared__ bf16 sVT[64][72];      // [d][key]     transposed V
  __shared__ bf16 sP[8][16][72];    // per-wave P tile, 16 x 64 keys

  const int bh = blockIdx.y;
  const int b = bh / heads, h = bh % heads;
  const int t = threadIdx.x, wave = t >> 5, lane = t & 31;
  const int half = lane >> 4, r = lane & 15;
  const int qs = blockIdx.x * 128 + wave * 16;
  const float scale = rsqrtf((float)dk);

  // Q fragments (row = lane%16, K-halves by lane/16), scale folded in.
  bf16x16 qF[2];
  {
    const float* qrow = Q + (size_t)(b * S + qs + r) * H + h * 64;
    const int kOff = half * 8;
#pragma unroll
    for (int c = 0; c < 2; ++c) {
      bf16x16 f;
#pragma unroll
      for (int i = 0; i < 8; ++i)
        f[i] = (bf16)(qrow[c * 32 + kOff + i] * scale);
#pragma unroll
      for (int i = 0; i < 8; ++i)
        f[i + 8] = (bf16)(qrow[c * 32 + 16 + kOff + i] * scale);
      qF[c] = f;
    }
  }

  floatx8 o[4];
  float mrun[8], lsum[8];
#pragma unroll
  for (int nd = 0; nd < 4; ++nd)
#pragma unroll
    for (int e = 0; e < 8; ++e) o[nd][e] = 0.f;
#pragma unroll
  for (int e = 0; e < 8; ++e) { mrun[e] = -3.0e38f; lsum[e] = 0.f; }

  const int lr = t >> 4;         // tile-stage row helper (16 rows/pass)
  const int lc = (t & 15) * 4;   // tile-stage col (float4)

  for (int kv0 = 0; kv0 < S; kv0 += 64) {
    // Cooperative K/V tile staging (fp32 -> bf16; V transposed on store).
#pragma unroll
    for (int p = 0; p < 4; ++p) {
      int trow = lr + p * 16;
      size_t grow = (size_t)(b * S + kv0 + trow) * H + h * 64 + lc;
      const float4 kf = *reinterpret_cast<const float4*>(Kv + grow);
      bf16x4 kb = {(bf16)kf.x, (bf16)kf.y, (bf16)kf.z, (bf16)kf.w};
      *reinterpret_cast<bf16x4*>(&sK[trow][lc]) = kb;
      const float4 vf = *reinterpret_cast<const float4*>(V + grow);
      sVT[lc + 0][trow] = (bf16)vf.x;
      sVT[lc + 1][trow] = (bf16)vf.y;
      sVT[lc + 2][trow] = (bf16)vf.z;
      sVT[lc + 3][trow] = (bf16)vf.w;
      if (kv0 + 64 < S) {  // L2 prefetch of next tile (global_prefetch_b8)
        __builtin_prefetch(Kv + grow + (size_t)64 * H, 0, 1);
        __builtin_prefetch(V + grow + (size_t)64 * H, 0, 1);
      }
    }
    __syncthreads();

    // Scores: S = (Q*scale) @ K^T  (B frag column = key, K-dim = d).
    floatx8 sc[4];
#pragma unroll
    for (int ni = 0; ni < 4; ++ni)
#pragma unroll
      for (int e = 0; e < 8; ++e) sc[ni][e] = 0.f;
#pragma unroll
    for (int c = 0; c < 2; ++c)
#pragma unroll
      for (int ni = 0; ni < 4; ++ni) {
        const bf16* rp = &sK[ni * 16 + r][0];
        bf16x16 kF = frag_from(rp + c * 32 + half * 8,
                               rp + c * 32 + 16 + half * 8);
        sc[ni] = WMMA_BF16(qF[c], kF, sc[ni]);
      }

    // Online softmax. Row e+half*8 lives across the 16 lanes of this half.
#pragma unroll
    for (int e = 0; e < 8; ++e) {
      float v = fmaxf(fmaxf(sc[0][e], sc[1][e]), fmaxf(sc[2][e], sc[3][e]));
#pragma unroll
      for (int mask = 1; mask < 16; mask <<= 1)
        v = fmaxf(v, __shfl_xor(v, mask, 32));
      float mnew = fmaxf(mrun[e], v);
      float alpha = __expf(mrun[e] - mnew);
      float rs = 0.f;
#pragma unroll
      for (int ni = 0; ni < 4; ++ni) {
        float p = __expf(sc[ni][e] - mnew);
        sc[ni][e] = p;
        rs += p;
      }
#pragma unroll
      for (int mask = 1; mask < 16; mask <<= 1) rs += __shfl_xor(rs, mask, 32);
      lsum[e] = lsum[e] * alpha + rs;
      mrun[e] = mnew;
#pragma unroll
      for (int nd = 0; nd < 4; ++nd) o[nd][e] *= alpha;
    }

    // P (C-layout) -> per-wave LDS tile (row-major) for A-fragment reload.
#pragma unroll
    for (int ni = 0; ni < 4; ++ni)
#pragma unroll
      for (int e = 0; e < 8; ++e)
        sP[wave][half * 8 + e][ni * 16 + r] = (bf16)sc[ni][e];
    // Same-wave LDS ops are in-order: store->load RAW is safe without barrier.

    // O += P @ V  (A = P over keys; B frag column = d via transposed V).
#pragma unroll
    for (int kc = 0; kc < 2; ++kc) {
      const bf16* pp = &sP[wave][r][0];
      bf16x16 pF = frag_from(pp + kc * 32 + half * 8,
                             pp + kc * 32 + 16 + half * 8);
#pragma unroll
      for (int nd = 0; nd < 4; ++nd) {
        const bf16* vp = &sVT[nd * 16 + r][0];
        bf16x16 vF = frag_from(vp + kc * 32 + half * 8,
                               vp + kc * 32 + 16 + half * 8);
        o[nd] = WMMA_BF16(pF, vF, o[nd]);
      }
    }
    __syncthreads();
  }

  // Normalize and write context (same [B,S,H] head-interleaved layout).
#pragma unroll
  for (int nd = 0; nd < 4; ++nd) {
    int col = h * 64 + nd * 16 + r;
#pragma unroll
    for (int e = 0; e < 8; ++e) {
      int srow = qs + half * 8 + e;
      O[(size_t)(b * S + srow) * H + col] = o[nd][e] / lsum[e];
    }
  }
}

// ---------------------------------------------------------------------------
// LayerNorm over last dim (H=1024), one row per block.
// ---------------------------------------------------------------------------
__global__ __launch_bounds__(256) void layernorm_kernel(
    const float* __restrict__ X, const float* __restrict__ g,
    const float* __restrict__ be, float* __restrict__ Y, int H) {
  const int row = blockIdx.x;
  const float* x = X + (size_t)row * H;
  float* y = Y + (size_t)row * H;
  float s = 0.f, s2 = 0.f;
  for (int i = threadIdx.x; i < H; i += blockDim.x) {
    float v = x[i];
    s += v;
    s2 += v * v;
  }
#pragma unroll
  for (int mask = 1; mask < 32; mask <<= 1) {
    s += __shfl_xor(s, mask, 32);
    s2 += __shfl_xor(s2, mask, 32);
  }
  __shared__ float wsum[8], wsq[8];
  int wave = threadIdx.x >> 5, lane = threadIdx.x & 31;
  if (lane == 0) { wsum[wave] = s; wsq[wave] = s2; }
  __syncthreads();
  if (wave == 0) {
    s = (lane < 8) ? wsum[lane] : 0.f;
    s2 = (lane < 8) ? wsq[lane] : 0.f;
#pragma unroll
    for (int mask = 1; mask < 8; mask <<= 1) {
      s += __shfl_xor(s, mask, 32);
      s2 += __shfl_xor(s2, mask, 32);
    }
    if (lane == 0) { wsum[0] = s; wsq[0] = s2; }
  }
  __syncthreads();
  float mu = wsum[0] / H;
  float var = wsq[0] / H - mu * mu;
  float rstd = rsqrtf(var + 1e-5f);
  for (int i = threadIdx.x; i < H; i += blockDim.x)
    y[i] = (x[i] - mu) * rstd * g[i] + be[i];
}

// ---------------------------------------------------------------------------
// Orchestration.
// ---------------------------------------------------------------------------
extern "C" void kernel_launch(void* const* d_in, const int* in_sizes, int n_in,
                              void* d_out, int out_size, void* d_ws,
                              size_t ws_size, hipStream_t stream) {
  (void)in_sizes; (void)n_in; (void)out_size; (void)ws_size;
  const int B = 4, S = 2048, H = 1024, HEADS = 16, DK = 64, DFF = 4096;
  const int M = B * S;

  const float* x = (const float*)d_in[0];
  const float* Wq = (const float*)d_in[1];
  const float* bq = (const float*)d_in[2];
  const float* Wk = (const float*)d_in[3];
  const float* bk = (const float*)d_in[4];
  const float* Wv = (const float*)d_in[5];
  const float* bv = (const float*)d_in[6];
  const float* Wo = (const float*)d_in[7];
  const float* bo = (const float*)d_in[8];
  const float* W1 = (const float*)d_in[9];
  const float* b1 = (const float*)d_in[10];
  const float* W2 = (const float*)d_in[11];
  const float* b2 = (const float*)d_in[12];
  const float* g1 = (const float*)d_in[13];
  const float* be1 = (const float*)d_in[14];
  const float* g2 = (const float*)d_in[15];
  const float* be2 = (const float*)d_in[16];

  char* ws = (char*)d_ws;
  size_t off = 0;
  auto alloc = [&](size_t bytes) -> char* {
    char* p = ws + off;
    off = (off + bytes + 255) & ~(size_t)255;
    return p;
  };
  float* q = (float*)alloc((size_t)M * H * 4);
  float* kk = (float*)alloc((size_t)M * H * 4);
  float* vv = (float*)alloc((size_t)M * H * 4);
  float* ctx = (float*)alloc((size_t)M * H * 4);
  float* y1 = (float*)alloc((size_t)M * H * 4);
  float* x1 = (float*)alloc((size_t)M * H * 4);
  float* h1 = (float*)alloc((size_t)M * DFF * 4);
  float* y2 = (float*)alloc((size_t)M * H * 4);
  bf16* wqT = (bf16*)alloc((size_t)H * H * 2);
  bf16* wkT = (bf16*)alloc((size_t)H * H * 2);
  bf16* wvT = (bf16*)alloc((size_t)H * H * 2);
  bf16* woT = (bf16*)alloc((size_t)H * H * 2);
  bf16* w1T = (bf16*)alloc((size_t)H * DFF * 2);
  bf16* w2T = (bf16*)alloc((size_t)DFF * H * 2);

  const int thr = 256;
  castT_f32_bf16<<<(H * H + thr - 1) / thr, thr, 0, stream>>>(Wq, wqT, H, H);
  castT_f32_bf16<<<(H * H + thr - 1) / thr, thr, 0, stream>>>(Wk, wkT, H, H);
  castT_f32_bf16<<<(H * H + thr - 1) / thr, thr, 0, stream>>>(Wv, wvT, H, H);
  castT_f32_bf16<<<(H * H + thr - 1) / thr, thr, 0, stream>>>(Wo, woT, H, H);
  castT_f32_bf16<<<(H * DFF + thr - 1) / thr, thr, 0, stream>>>(W1, w1T, H, DFF);
  castT_f32_bf16<<<(H * DFF + thr - 1) / thr, thr, 0, stream>>>(W2, w2T, DFF, H);

  dim3 gproj(H / 128, M / 128);
  gemm_bf16_wmma<false, false><<<gproj, 256, 0, stream>>>(x, wqT, bq, nullptr, q, M, H, H);
  gemm_bf16_wmma<false, false><<<gproj, 256, 0, stream>>>(x, wkT, bk, nullptr, kk, M, H, H);
  gemm_bf16_wmma<false, false><<<gproj, 256, 0, stream>>>(x, wvT, bv, nullptr, vv, M, H, H);

  flash_attn<<<dim3(S / 128, B * HEADS), 256, 0, stream>>>(q, kk, vv, ctx, B, S, H, HEADS, DK);

  // attn_out + residual(x), then LN1
  gemm_bf16_wmma<false, true><<<gproj, 256, 0, stream>>>(ctx, woT, bo, x, y1, M, H, H);
  layernorm_kernel<<<M, 256, 0, stream>>>(y1, g1, be1, x1, H);

  // FFN: GELU(x1 @ W1 + b1) @ W2 + b2 + x1, then LN2 -> d_out
  gemm_bf16_wmma<true, false><<<dim3(DFF / 128, M / 128), 256, 0, stream>>>(x1, w1T, b1, nullptr, h1, M, DFF, H);
  gemm_bf16_wmma<false, true><<<gproj, 256, 0, stream>>>(h1, w2T, b2, x1, y2, M, H, DFF);
  layernorm_kernel<<<M, 256, 0, stream>>>(y2, g2, be2, (float*)d_out, H);
}
